// MambaBlock_515396076431
// MI455X (gfx1250) — matrix operand order repaired and checked by reading, not compiled, not run on hardware
//
#include <hip/hip_runtime.h>
#include <hip/hip_bf16.h>

typedef __attribute__((ext_vector_type(16))) _Float16 v16h_t;
typedef __attribute__((ext_vector_type(8)))  _Float16 v8h_t;
typedef __attribute__((ext_vector_type(4)))  _Float16 v4h_t;
typedef __attribute__((ext_vector_type(8)))  float    v8f_t;
typedef int v4i_vs __attribute__((vector_size(16)));   // matches builtin param

#define D_MODEL 1024
#define D_INNER 2048
#define NSTATE  16
#define DT_RANK 64
#define SEQLEN  2048
#define NBATCH  2
#define BL      (NBATCH * SEQLEN)       // 4096 rows
#define XPROJ_N (DT_RANK + 2 * NSTATE)  // 96
#define EPS_F   1.1920928955078125e-07f

#if defined(__has_builtin)
#if __has_builtin(__builtin_amdgcn_global_load_async_to_lds_b128)
#define USE_ASYNC_LDS 1
#endif
#endif

__device__ __forceinline__ void async_b128(const void* g, void* l) {
#ifdef USE_ASYNC_LDS
  __builtin_amdgcn_global_load_async_to_lds_b128(
      (__attribute__((address_space(1))) v4i_vs*)g,
      (__attribute__((address_space(3))) v4i_vs*)l, 0, 0);
#else
  *(v8h_t*)l = *(const v8h_t*)g;
#endif
}

__device__ __forceinline__ void wait_async0() {
#ifdef USE_ASYNC_LDS
#if __has_builtin(__builtin_amdgcn_s_wait_asynccnt)
  __builtin_amdgcn_s_wait_asynccnt(0);
#else
  asm volatile("s_wait_asynccnt 0" ::: "memory");
#endif
#endif
}

// ---------------------------------------------------------------------------
// RMSNorm: one block per row of 1024; writes f16 (GEMM input)
// ---------------------------------------------------------------------------
__global__ __launch_bounds__(256) void rmsnorm_kernel(
    const float* __restrict__ x, const float* __restrict__ w,
    _Float16* __restrict__ h) {
  __shared__ float red[256];
  int row = blockIdx.x;
  const float* xr = x + (size_t)row * D_MODEL;
  float s = 0.f;
  for (int i = threadIdx.x; i < D_MODEL; i += 256) {
    float v = xr[i];
    s += v * v;
  }
  red[threadIdx.x] = s;
  __syncthreads();
  for (int off = 128; off > 0; off >>= 1) {
    if (threadIdx.x < off) red[threadIdx.x] += red[threadIdx.x + off];
    __syncthreads();
  }
  float r = rsqrtf(red[0] * (1.0f / D_MODEL) + EPS_F);
  _Float16* hr = h + (size_t)row * D_MODEL;
  for (int i = threadIdx.x; i < D_MODEL; i += 256)
    hr[i] = (_Float16)(xr[i] * r * w[i]);
}

// ---------------------------------------------------------------------------
// fp32 -> f16 conversion (weights), 4 elems/thread
// ---------------------------------------------------------------------------
__global__ __launch_bounds__(256) void f32_to_f16_kernel(
    const float* __restrict__ in, _Float16* __restrict__ out) {
  int g = blockIdx.x * 256 + threadIdx.x;
  float4 v = ((const float4*)in)[g];
  v4h_t h;
  h[0] = (_Float16)v.x; h[1] = (_Float16)v.y;
  h[2] = (_Float16)v.z; h[3] = (_Float16)v.w;
  ((v4h_t*)out)[g] = h;
}

// ---------------------------------------------------------------------------
// WMMA GEMM: C(M,N) = A(M,K;lda) * W(N,K)^T, f16 operands, fp32 accumulate.
// 128x128 block tile, K-step 32, 8 waves, each wave 32x64 (2x4 wmma frags).
// Double-buffered LDS; tiles staged with async global->LDS copies (ASYNCcnt)
// so tile k+1 copies overlap tile k WMMAs with no VGPR staging.
// MODE 0: f32 store   MODE 1: softplus(val + extra[col])
// MODE 2: val + extra[row*ldc+col] (residual)   MODE 3: f32 + f16 dual store
// GUARD: clamp out-of-range rows/cols on load (garbage cols never stored)
// ---------------------------------------------------------------------------
template <int MODE, bool GUARD>
__global__ __launch_bounds__(256) void gemm_wmma_f16(
    const _Float16* __restrict__ A, int lda,
    const _Float16* __restrict__ W,           // (N, K) row-major
    float* __restrict__ C, _Float16* __restrict__ C16, int ldc,
    int M, int N, int K,
    const float* __restrict__ extra) {
  __shared__ _Float16 As[2][128][40];         // +8 halves pad, rows 16B-aligned
  __shared__ _Float16 Bs[2][128][40];

  const int tid  = threadIdx.x;
  const int rowBase = blockIdx.y * 128;
  const int colBase = blockIdx.x * 128;
  const int wave = tid >> 5;
  const int lane = tid & 31;
  const int wr = (wave & 3) * 32;             // wave row offset in tile
  const int wc = (wave >> 2) * 64;            // wave col offset in tile
  const int m16 = lane & 15;
  const int hi  = lane >> 4;                  // lane group 0/1

  // staging: thread t copies 16B chunks (8 halves) at rows r0, r0+64
  const int r0 = tid >> 2;                    // 0..63
  const int kc = (tid & 3) * 8;               // 0,8,16,24

  v8f_t acc[2][4];
#pragma unroll
  for (int i = 0; i < 2; ++i)
#pragma unroll
    for (int j = 0; j < 4; ++j) acc[i][j] = (v8f_t)0.f;

  auto stage = [&](int k0, int buf) {
#pragma unroll
    for (int p = 0; p < 2; ++p) {
      int row = r0 + p * 64;
      int gr = rowBase + row;
      int gc = colBase + row;
      if (GUARD) {
        gr = (gr < M) ? gr : (M - 1);
        gc = (gc < N) ? gc : (N - 1);
      }
      async_b128(A + (size_t)gr * lda + k0 + kc, &As[buf][row][kc]);
      async_b128(W + (size_t)gc * K + k0 + kc, &Bs[buf][row][kc]);
    }
  };

  stage(0, 0);
  wait_async0();
  __syncthreads();

  int buf = 0;
  for (int k0 = 0; k0 < K; k0 += 32) {
    const bool more = (k0 + 32 < K);
    if (more) stage(k0 + 32, buf ^ 1);        // overlaps WMMA below

    _Float16 (*Asb)[40] = As[buf];
    _Float16 (*Bsb)[40] = Bs[buf];

    // A fragment: lane<16 -> K[0..7],[16..23]; lane>=16 -> K[8..15],[24..31]
    const int klo = hi * 8;
    v16h_t af[2];
#pragma unroll
    for (int i = 0; i < 2; ++i) {
      const v8h_t* pa = (const v8h_t*)&Asb[wr + i * 16 + m16][klo];
      v8h_t a0 = pa[0];     // K klo .. klo+7
      v8h_t a1 = pa[2];     // K klo+16 .. klo+23
#pragma unroll
      for (int e = 0; e < 8; ++e) { af[i][e] = a0[e]; af[i][e + 8] = a1[e]; }
    }
    // B fragment: lane gives N=m16; lane<16 -> K[0..7],[16..23]; lane>=16 -> K[8..15],[24..31]
    const int kb = hi * 8;
    v16h_t bf[4];
#pragma unroll
    for (int j = 0; j < 4; ++j) {
      const v8h_t* pb = (const v8h_t*)&Bsb[wc + j * 16 + m16][kb];
      v8h_t b0 = pb[0];
      v8h_t b1 = pb[2];
#pragma unroll
      for (int e = 0; e < 8; ++e) { bf[j][e] = b0[e]; bf[j][e + 8] = b1[e]; }
    }
#pragma unroll
    for (int i = 0; i < 2; ++i)
#pragma unroll
      for (int j = 0; j < 4; ++j)
        acc[i][j] = __builtin_amdgcn_wmma_f32_16x16x32_f16(
            false, af[i], false, bf[j], (short)0, acc[i][j], false, false);

    if (more) {
      wait_async0();                          // my copies into buf^1 done
      __syncthreads();                        // everyone's copies + reads done
    }
    buf ^= 1;
  }

  // C/D layout: VGPR v -> row = hi*8 + v, col = m16
#pragma unroll
  for (int i = 0; i < 2; ++i) {
#pragma unroll
    for (int j = 0; j < 4; ++j) {
      int col = colBase + wc + j * 16 + m16;
      if (GUARD && col >= N) continue;
#pragma unroll
      for (int v = 0; v < 8; ++v) {
        int row = rowBase + wr + i * 16 + hi * 8 + v;
        if (GUARD && row >= M) continue;
        float val = acc[i][j][v];
        if (MODE == 1) {
          val += extra[col];
          val = (val > 20.f) ? val : __logf(1.f + __expf(val));
        } else if (MODE == 2) {
          val += extra[(size_t)row * ldc + col];
        }
        C[(size_t)row * ldc + col] = val;
        if (MODE == 3) C16[(size_t)row * ldc + col] = (_Float16)val;
      }
    }
  }
}

// ---------------------------------------------------------------------------
// Causal depthwise conv1d (K=4) + SiLU.  u = xz[:, 0:2048]
// Writes f32 (scan input) and f16 (x_proj GEMM input).
// ---------------------------------------------------------------------------
__global__ __launch_bounds__(256) void conv_silu_kernel(
    const float* __restrict__ xz, const float* __restrict__ cw,
    const float* __restrict__ cb, float* __restrict__ uc32,
    _Float16* __restrict__ uc16) {
  int g = blockIdx.x * 256 + threadIdx.x;       // 0 .. BL*D_INNER-1
  int d = g & (D_INNER - 1);
  int r = g >> 11;                              // row in [0, BL)
  int l = r & (SEQLEN - 1);
  float acc = cb[d];
#pragma unroll
  for (int k = 0; k < 4; ++k) {
    int li = l - 3 + k;
    if (li >= 0)
      acc += xz[(size_t)(r - 3 + k) * (2 * D_INNER) + d] * cw[d * 4 + k];
  }
  float s = acc / (1.f + __expf(-acc));
  uc32[g] = s;
  uc16[g] = (_Float16)s;
}

// ---------------------------------------------------------------------------
// Selective scan. One (b,d) channel per 16 lanes (n = lane%16), sequential
// over L. y = (sum_n h*C + D*uc) * silu(z); writes f16 (out_proj GEMM input).
// ---------------------------------------------------------------------------
__global__ __launch_bounds__(256) void scan_kernel(
    const float* __restrict__ dt,   const float* __restrict__ uc,
    const float* __restrict__ xdbl, const float* __restrict__ xz,
    const float* __restrict__ A_log, const float* __restrict__ Dp,
    _Float16* __restrict__ y) {
  int tid = threadIdx.x;
  int n  = tid & 15;
  int ch = blockIdx.x * 16 + (tid >> 4);        // 0 .. NBATCH*D_INNER-1
  int b  = ch / D_INNER;
  int d  = ch % D_INNER;
  float Aval = -__expf(A_log[d * NSTATE + n]);
  float Dval = Dp[d];
  float hstate = 0.f;
  const float* dtp = dt + (size_t)b * SEQLEN * D_INNER + d;
  const float* ucp = uc + (size_t)b * SEQLEN * D_INNER + d;
  const float* zp  = xz + (size_t)b * SEQLEN * (2 * D_INNER) + D_INNER + d;
  const float* Bp  = xdbl + (size_t)b * SEQLEN * XPROJ_N + DT_RANK + n;
  const float* Cp  = Bp + NSTATE;
  _Float16* yp = y + (size_t)b * SEQLEN * D_INNER + d;
#pragma unroll 2
  for (int l = 0; l < SEQLEN; ++l) {
    float dtv = dtp[(size_t)l * D_INNER];
    float ucv = ucp[(size_t)l * D_INNER];
    float Bn  = Bp[(size_t)l * XPROJ_N];
    float Cn  = Cp[(size_t)l * XPROJ_N];
    float dA = __expf(dtv * Aval);
    hstate = dA * hstate + dtv * Bn * ucv;
    float p = hstate * Cn;
    p += __shfl_xor(p, 1, 32);
    p += __shfl_xor(p, 2, 32);
    p += __shfl_xor(p, 4, 32);
    p += __shfl_xor(p, 8, 32);
    if (n == 0) {
      float yv = p + Dval * ucv;
      float zv = zp[(size_t)l * (2 * D_INNER)];
      float sz = zv / (1.f + __expf(-zv));
      yp[(size_t)l * D_INNER] = (_Float16)(yv * sz);
    }
  }
}

// ---------------------------------------------------------------------------
extern "C" void kernel_launch(void* const* d_in, const int* in_sizes, int n_in,
                              void* d_out, int out_size, void* d_ws, size_t ws_size,
                              hipStream_t stream) {
  const float* x          = (const float*)d_in[0];
  const float* norm_w     = (const float*)d_in[1];
  const float* in_proj_w  = (const float*)d_in[2];   // (4096, 1024)
  const float* conv_w     = (const float*)d_in[3];   // (2048, 4)
  const float* conv_b     = (const float*)d_in[4];
  const float* x_proj_w   = (const float*)d_in[5];   // (96, 2048)
  const float* dt_proj_w  = (const float*)d_in[6];   // (2048, 64)
  const float* dt_proj_b  = (const float*)d_in[7];
  const float* A_log      = (const float*)d_in[8];   // (2048, 16)
  const float* Dp         = (const float*)d_in[9];
  const float* out_proj_w = (const float*)d_in[10];  // (1024, 2048)
  float* out = (float*)d_out;

  // fp32 scratch
  float* wsf    = (float*)d_ws;
  float* xz     = wsf;                  // 16777216
  float* uc32   = wsf + 16777216;       //  8388608
  float* xdbl32 = wsf + 25165824;       //   393216
  float* dt     = wsf + 25559040;       //  8388608
  // f16 scratch (starts at float offset 33947648 -> 16B aligned)
  _Float16* wsh    = (_Float16*)(wsf + 33947648);
  _Float16* h16    = wsh;               //  4194304
  _Float16* w_in16 = wsh + 4194304;     //  4194304
  _Float16* w_x16  = wsh + 8388608;     //   196608
  _Float16* w_dt16 = wsh + 8585216;     //   131072
  _Float16* w_out16= wsh + 8716288;     //  2097152
  _Float16* uc16   = wsh + 10813440;    //  8388608
  _Float16* xdbl16 = wsh + 19202048;    //   393216
  _Float16* y16    = wsh + 19595264;    //  8388608

  // 0. one-time weight conversions to f16 (each matrix reused 8-32x by GEMMs)
  f32_to_f16_kernel<<<4096, 256, 0, stream>>>(in_proj_w,  w_in16);
  f32_to_f16_kernel<<<192,  256, 0, stream>>>(x_proj_w,   w_x16);
  f32_to_f16_kernel<<<128,  256, 0, stream>>>(dt_proj_w,  w_dt16);
  f32_to_f16_kernel<<<2048, 256, 0, stream>>>(out_proj_w, w_out16);

  // 1. RMSNorm -> h16
  rmsnorm_kernel<<<BL, 256, 0, stream>>>(x, norm_w, h16);

  // 2. in_proj: xz = h @ in_proj_w.T   (4096 x 4096, K=1024)
  gemm_wmma_f16<0, false><<<dim3(32, 32), 256, 0, stream>>>(
      h16, D_MODEL, w_in16, xz, nullptr, 2 * D_INNER, BL, 2 * D_INNER, D_MODEL,
      nullptr);

  // 3. causal conv + SiLU -> uc (f32 + f16)
  conv_silu_kernel<<<(BL * D_INNER) / 256, 256, 0, stream>>>(
      xz, conv_w, conv_b, uc32, uc16);

  // 4. x_proj: x_dbl = uc @ x_proj_w.T  (4096 x 96, K=2048), dual f32/f16 out
  gemm_wmma_f16<3, true><<<dim3(1, 32), 256, 0, stream>>>(
      uc16, D_INNER, w_x16, xdbl32, xdbl16, XPROJ_N, BL, XPROJ_N, D_INNER,
      nullptr);

  // 5. dt_proj + bias + softplus: dt (4096 x 2048, K=64)
  gemm_wmma_f16<1, false><<<dim3(16, 32), 256, 0, stream>>>(
      xdbl16, XPROJ_N, w_dt16, dt, nullptr, D_INNER, BL, D_INNER, DT_RANK,
      dt_proj_b);

  // 6. selective scan + D-skip + SiLU(z) gate -> y16 (4096 x 2048)
  scan_kernel<<<(NBATCH * D_INNER) / 16, 256, 0, stream>>>(
      dt, uc32, xdbl32, xz, A_log, Dp, y16);

  // 7. out_proj + residual: out = y @ out_proj_w.T + x  (4096 x 1024, K=2048)
  gemm_wmma_f16<2, false><<<dim3(8, 32), 256, 0, stream>>>(
      y16, D_INNER, w_out16, out, nullptr, D_MODEL, BL, D_MODEL, D_INNER, x);
}
